// Quantizer_59777354826461
// MI455X (gfx1250) — matrix operand order repaired and checked by reading, not compiled, not run on hardware
//
#include <hip/hip_runtime.h>

typedef __attribute__((ext_vector_type(16))) _Float16 v16h;
typedef __attribute__((ext_vector_type(8)))  float    v8f;
typedef int   v4i __attribute__((vector_size(16)));
typedef float f4v __attribute__((ext_vector_type(4)));

#define B_  16
#define D_  512
#define T_  4096
#define K_  1024
#define N_  (B_ * T_)            // 65536 rows
#define NKT (K_ / 16)            // 64 code tiles
#define NDT (D_ / 32)            // 16 k-dim tiles

// Async global->LDS copy (CDNA5 GLOBAL_LOAD_ASYNC_TO_LDS_B128, ASYNCcnt),
// with a register-staging fallback if the builtins are unavailable.
#if defined(__has_builtin)
#if __has_builtin(__builtin_amdgcn_global_load_async_to_lds_b128) && \
    __has_builtin(__builtin_amdgcn_s_wait_asynccnt)
#define USE_ASYNC_LDS 1
#endif
#endif
#ifndef USE_ASYNC_LDS
#define USE_ASYNC_LDS 0
#endif

#if USE_ASYNC_LDS
__device__ __forceinline__ void async_cp16(const void* g, void* l) {
    __builtin_amdgcn_global_load_async_to_lds_b128(
        (__attribute__((address_space(1))) v4i*)g,
        (__attribute__((address_space(3))) v4i*)l, 0, 0);
}
#endif

// ---------------------------------------------------------------------------
// Kernel 1: embedding f32 -> f16 in the WMMA B-fragment layout used here (contiguous k).
//   bfrag[((kt*16+dt)*32 + lane)*16 + i] = e[kt*16+(lane&15)][dt*32 + i + 16*(lane>=16)]
// Also zeroes the loss accumulator (runs every launch, before vq_main).
// ---------------------------------------------------------------------------
__global__ void vq_convert_embed(const float* __restrict__ e,
                                 _Float16* __restrict__ bfrag,
                                 float* __restrict__ lossAcc) {
    int gid  = blockIdx.x * blockDim.x + threadIdx.x;   // 0 .. 32767
    if (gid == 0) *lossAcc = 0.0f;
    int lane = gid & 31;
    int frag = gid >> 5;                                // kt*16 + dt
    int kt = frag >> 4, dt = frag & 15;
    int code = kt * 16 + (lane & 15);
    int d0   = dt * 32 + ((lane >> 4) << 4);
    const float* src = e + (size_t)code * D_ + d0;
    v16h tmp;
#pragma unroll
    for (int i = 0; i < 16; ++i) tmp[i] = (_Float16)src[i];
    *(v16h*)(bfrag + (size_t)gid * 16) = tmp;
}

// ---------------------------------------------------------------------------
// Kernel 2: e2[k] = sum_d e[k][d]^2   (one wave32 per codebook row)
// ---------------------------------------------------------------------------
__global__ void vq_embed_norms(const float* __restrict__ e,
                               float* __restrict__ e2) {
    int wave = (blockIdx.x * blockDim.x + threadIdx.x) >> 5;  // 0..1023
    int lane = threadIdx.x & 31;
    const float* row = e + (size_t)wave * D_ + lane * 16;
    float s = 0.0f;
#pragma unroll
    for (int i = 0; i < 16; ++i) { float v = row[i]; s += v * v; }
#pragma unroll
    for (int m = 16; m >= 1; m >>= 1) s += __shfl_xor(s, m, 32);
    if (lane == 0) e2[wave] = s;
}

// ---------------------------------------------------------------------------
// Kernel 3 (main): per wave = 16 rows (consecutive t within one b).
//   A (x-tile, f16) resident in 128 VGPRs; B tiles double-buffered through
//   LDS via async copies (shared by all 8 waves); running score-min/argmin
//   in registers; epilogue: cross-lane min reduce, loss atomic, row gather.
// ---------------------------------------------------------------------------
__global__ void __launch_bounds__(256)
vq_main(const float* __restrict__ x, const float* __restrict__ e,
        const _Float16* __restrict__ bfrag, const float* __restrict__ e2,
        float* __restrict__ outq, float* __restrict__ outidx,
        float* __restrict__ lossAcc) {
    __shared__ _Float16 ldsB[2][16 * 512];   // 2 x 16 KB double buffer
    __shared__ int      ldsIdx[8][16];

    const int tid   = threadIdx.x;
    const int wave  = tid >> 5;
    const int lane  = tid & 31;
    const int half  = lane >> 4;          // 0: K-low, 1: K-high half of frag
    const int r     = lane & 15;          // A row within tile / B column
    const int nBase = blockIdx.x * 128 + wave * 16;
    const int b     = nBase >> 12;        // / T_
    const int tBase = nBase & (T_ - 1);

    // ---- load A fragments (x tile, streamed once -> nontemporal),
    //      accumulate sum(x^2) on the fly ----
    const float* xb = x + (size_t)b * ((size_t)D_ * T_) + tBase + r;
    v16h  a[NDT];
    float xsq = 0.0f;
#pragma unroll
    for (int dt = 0; dt < NDT; ++dt) {
#pragma unroll
        for (int i = 0; i < 16; ++i) {
            int d = dt * 32 + i + (half << 4);
            float v = __builtin_nontemporal_load(&xb[(size_t)d * T_]);
            xsq += v * v;
            a[dt][i] = (_Float16)v;
        }
    }

    float minv[8];
    int   mini[8];
#pragma unroll
    for (int v = 0; v < 8; ++v) { minv[v] = 3.4e38f; mini[v] = 0; }

    // ---- prime buffer 0 with code tile 0 ----
#if USE_ASYNC_LDS
    {
        const char* src = (const char*)bfrag;
        char*       dst = (char*)&ldsB[0][0];
#pragma unroll
        for (int j = 0; j < 4; ++j)
            async_cp16(src + (tid + j * 256) * 16, dst + (tid + j * 256) * 16);
        __builtin_amdgcn_s_wait_asynccnt(0);
    }
#else
    {
        const uint4* src = (const uint4*)bfrag;
        uint4*       dst = (uint4*)&ldsB[0][0];
#pragma unroll
        for (int j = 0; j < 4; ++j) dst[tid + j * 256] = src[tid + j * 256];
    }
#endif
    __syncthreads();

    // ---- main loop over 64 code tiles (double-buffered, 1 barrier/tile) ----
    for (int kt = 0; kt < NKT; ++kt) {
        const int buf = kt & 1;

        // issue next tile's copy early so it hides under the WMMAs
#if USE_ASYNC_LDS
        if (kt + 1 < NKT) {
            const char* src = (const char*)(bfrag + (size_t)(kt + 1) * 8192);
            char*       dst = (char*)&ldsB[buf ^ 1][0];
#pragma unroll
            for (int j = 0; j < 4; ++j)
                async_cp16(src + (tid + j * 256) * 16, dst + (tid + j * 256) * 16);
        }
#else
        uint4 stage[4];
        if (kt + 1 < NKT) {
            const uint4* src = (const uint4*)(bfrag + (size_t)(kt + 1) * 8192);
#pragma unroll
            for (int j = 0; j < 4; ++j) stage[j] = src[tid + j * 256];
        }
#endif

        float e2v = e2[kt * 16 + r];
        v8f acc = {};
#pragma unroll
        for (int dt = 0; dt < NDT; ++dt) {
            v16h bf = *(const v16h*)&ldsB[buf][dt * 512 + lane * 16];
            acc = __builtin_amdgcn_wmma_f32_16x16x32_f16(
                false, a[dt], false, bf, (short)0, acc, false, false);
        }
        int code = kt * 16 + r;
#pragma unroll
        for (int v = 0; v < 8; ++v) {
            float score = e2v - 2.0f * acc[v];   // ||e||^2 - 2 x.e
            if (score < minv[v]) { minv[v] = score; mini[v] = code; }
        }

#if !USE_ASYNC_LDS
        if (kt + 1 < NKT) {
            uint4* dst = (uint4*)&ldsB[buf ^ 1][0];
#pragma unroll
            for (int j = 0; j < 4; ++j) dst[tid + j * 256] = stage[j];
        }
#else
        __builtin_amdgcn_s_wait_asynccnt(0);
#endif
        __syncthreads();
    }

    // ---- reduce argmin across the 16 lanes of each half ----
    float msum = 0.0f;
#pragma unroll
    for (int v = 0; v < 8; ++v) {
        float mv = minv[v]; int mi = mini[v];
#pragma unroll
        for (int m = 8; m >= 1; m >>= 1) {
            float ov = __shfl_xor(mv, m, 32);
            int   oi = __shfl_xor(mi, m, 32);
            if (ov < mv || (ov == mv && oi < mi)) { mv = ov; mi = oi; }
        }
        msum += mv;                                // per-row min score
        if (r == 0) ldsIdx[wave][v + half * 8] = mi;
    }

    // loss contribution: sum over rows of (min_score + ||x||^2)
    float lossLocal = xsq + ((r == 0) ? msum : 0.0f);
#pragma unroll
    for (int m = 16; m >= 1; m >>= 1) lossLocal += __shfl_xor(lossLocal, m, 32);
    if (lane == 0) atomicAdd(lossAcc, lossLocal);

    __syncthreads();   // make ldsIdx visible

    // ---- gather chosen code rows -> quantized_st (nontemporal f32x4) ----
#pragma unroll
    for (int rr = 0; rr < 16; ++rr) {
        int idx = ldsIdx[wave][rr];
        const f4v* src = (const f4v*)(e + (size_t)idx * D_);
        f4v*       dst = (f4v*)(outq + (size_t)(nBase + rr) * D_);
#pragma unroll
        for (int j = 0; j < 4; ++j) {
            f4v val = src[lane + j * 32];
            __builtin_nontemporal_store(val, &dst[lane + j * 32]);
        }
    }
    if (lane < 16) outidx[nBase + lane] = (float)ldsIdx[wave][lane];
}

// ---------------------------------------------------------------------------
// Kernel 4: finalize loss = 1.25 * sum / (N*D)
// ---------------------------------------------------------------------------
__global__ void vq_finalize(const float* __restrict__ lossAcc,
                            float* __restrict__ lossOut) {
    *lossOut = 1.25f * (*lossAcc) * (1.0f / 33554432.0f);  // 1/(N*D)
}

// ---------------------------------------------------------------------------
extern "C" void kernel_launch(void* const* d_in, const int* in_sizes, int n_in,
                              void* d_out, int out_size, void* d_ws, size_t ws_size,
                              hipStream_t stream) {
    const float* x = (const float*)d_in[0];     // [B, D, T] f32
    const float* e = (const float*)d_in[1];     // [K, D]    f32

    float* outq   = (float*)d_out;                       // [N*D]
    float* lossO  = (float*)d_out + (size_t)N_ * D_;     // [1]
    float* outidx = lossO + 1;                           // [N]

    // workspace layout
    _Float16* bfrag   = (_Float16*)d_ws;                              // 1 MB
    float*    e2      = (float*)((char*)d_ws + (size_t)K_ * D_ * 2);  // 4 KB
    float*    lossAcc = e2 + K_;                                      // 4 B

    vq_convert_embed<<<128, 256, 0, stream>>>(e, bfrag, lossAcc);
    vq_embed_norms<<<128, 256, 0, stream>>>(e, e2);
    vq_main<<<N_ / 128, 256, 0, stream>>>(x, e, bfrag, e2, outq, outidx, lossAcc);
    vq_finalize<<<1, 1, 0, stream>>>(lossAcc, lossO);
}
